// GATClassifier_36618891165997
// MI455X (gfx1250) — matrix operand order repaired and checked by reading, not compiled, not run on hardware
//
#include <hip/hip_runtime.h>
#include <hip/hip_bf16.h>
#include <stdint.h>

// Problem constants (match reference)
#define NNODES 50000
#define NEDGES 800000
#define ETOT   (NEDGES + NNODES)   // self loops appended
#define INC    128
#define HEADS  4
#define CH     64
#define HC     (HEADS*CH)          // 256
#define NGRAPH 64
#define SLOPE  0.2f

typedef __attribute__((ext_vector_type(16))) _Float16 v16h_t;
typedef __attribute__((ext_vector_type(8)))  _Float16 v8h_t;
typedef __attribute__((ext_vector_type(8)))  float    v8f_t;

__device__ __forceinline__ unsigned f2ord(float f) {
  unsigned u = __float_as_uint(f);
  return (u & 0x80000000u) ? ~u : (u | 0x80000000u);
}
__device__ __forceinline__ float ord2f(unsigned u) {
  u = (u & 0x80000000u) ? (u & 0x7FFFFFFFu) : ~u;
  return __uint_as_float(u);
}

// ---------- generic fill ----------
__global__ void k_fill_u32(unsigned* __restrict__ p, unsigned v, long long n) {
  long long i = (long long)blockIdx.x * blockDim.x + threadIdx.x;
  if (i < n) p[i] = v;
}

// ---------- f32 -> f16 convert ----------
__global__ void k_f32_to_f16(_Float16* __restrict__ dst, const float* __restrict__ src,
                             long long n) {
  long long i = (long long)blockIdx.x * blockDim.x + threadIdx.x;
  if (i < n) dst[i] = (_Float16)src[i];
}

// ---------- pack W [K, Nout] row-major -> wave32 WMMA B-fragment layout ----------
// Bpack layout: [nt][kt][lane(32)][e(16)] halfs; lane = column (n = nt*16 + lane&15),
// lane<16 holds K offsets 0..15 of the k-step, lane>=16 holds 16..31; element e = K offset.
__global__ void k_pack_w(_Float16* __restrict__ bp, const float* __restrict__ w,
                         int K, int Nout) {
  int idx = blockIdx.x * blockDim.x + threadIdx.x;
  int kt_cnt = K >> 5;
  int total = (Nout >> 4) * kt_cnt * 32 * 16;
  if (idx >= total) return;
  int e    = idx & 15;
  int lane = (idx >> 4) & 31;
  int rest = idx >> 9;
  int kt   = rest % kt_cnt;
  int nt   = rest / kt_cnt;
  int k = kt * 32 + ((lane >> 4) << 4) + e;
  int n = nt * 16 + (lane & 15);
  bp[idx] = (_Float16)w[(size_t)k * Nout + n];
}

// ---------- WMMA GEMM: C[M,Nout] = A[M,K](f16) * Bpack(f16), f32 accumulate ----------
__global__ void k_gemm_wmma(const _Float16* __restrict__ A,
                            const _Float16* __restrict__ Bp,
                            float* __restrict__ C,
                            int M, int K, int Nout) {
  int gtid = blockIdx.x * blockDim.x + threadIdx.x;
  int wave = gtid >> 5;
  int lane = gtid & 31;
  int ntiles = Nout >> 4;
  int mt = wave / ntiles;
  int nt = wave - mt * ntiles;
  if (mt >= (M >> 4)) return;

  int m  = mt * 16 + (lane & 15);
  int ko = (lane < 16) ? 0 : 16;                // A-fragment K phase per lane half
  const _Float16* arow  = A + (size_t)m * K;
  const _Float16* bbase = Bp + ((size_t)nt * (K >> 5)) * 512 + lane * 16;

  v8f_t c = {};
  for (int k0 = 0; k0 < K; k0 += 32) {
    union { v16h_t v; v8h_t h[2]; } a, b;
    a.h[0] = *(const v8h_t*)(arow + k0 + ko);        // K = k0+ko .. +7
    a.h[1] = *(const v8h_t*)(arow + k0 + ko + 8);    // K = k0+ko+8 .. +15
    b.v    = *(const v16h_t*)(bbase);
    __builtin_prefetch((const void*)(bbase + 512), 0, 1);  // global_prefetch_b8
    bbase += 512;
    c = __builtin_amdgcn_wmma_f32_16x16x32_f16(
        /*neg_a=*/false, a.v, /*neg_b=*/false, b.v,
        /*c_mod=*/(short)0, c, /*reuse_a=*/false, /*reuse_b=*/false);
  }

  // C/D layout: VGPR r -> row (lane<16 ? r : r+8), col = lane&15
  int nbase = nt * 16 + (lane & 15);
  int mrow  = mt * 16 + ((lane < 16) ? 0 : 8);
#pragma unroll
  for (int r = 0; r < 8; ++r)
    C[(size_t)(mrow + r) * Nout + nbase] = c[r];
}

// ---------- per-node attention logits: as[n,h]=<h_lin[n,h,:],a_src[h,:]>, same for ad ----
__global__ void k_alphas(const float* __restrict__ hlin,
                         const float* __restrict__ a_src,
                         const float* __restrict__ a_dst,
                         float* __restrict__ as, float* __restrict__ ad, int n) {
  int i = blockIdx.x * blockDim.x + threadIdx.x;   // (node, head)
  if (i >= n * HEADS) return;
  int node = i >> 2, h = i & 3;
  const float* hp = hlin + (size_t)node * HC + h * CH;
  const float* s  = a_src + h * CH;
  const float* d  = a_dst + h * CH;
  float accs = 0.f, accd = 0.f;
#pragma unroll 4
  for (int c = 0; c < CH; ++c) { float v = hp[c]; accs += v * s[c]; accd += v * d[c]; }
  as[i] = accs; ad[i] = accd;
}

// ---------- edge pass 1: segment max (ordered-uint atomicMax) ----------
__global__ void k_edge_max(const int* __restrict__ ei,
                           const float* __restrict__ as, const float* __restrict__ ad,
                           unsigned* __restrict__ maxu) {
  int e = blockIdx.x * blockDim.x + threadIdx.x;
  if (e >= ETOT) return;
  int s = (e < NEDGES) ? ei[e]          : (e - NEDGES);
  int d = (e < NEDGES) ? ei[NEDGES + e] : (e - NEDGES);
  const float4 vs = *(const float4*)(as + s * 4);
  const float4 vd = *(const float4*)(ad + d * 4);
  float v[4] = { vs.x + vd.x, vs.y + vd.y, vs.z + vd.z, vs.w + vd.w };
#pragma unroll
  for (int h = 0; h < 4; ++h) {
    float lv = v[h] > 0.f ? v[h] : SLOPE * v[h];
    atomicMax(&maxu[d * 4 + h], f2ord(lv));
  }
}

// ---------- edge pass 2: ex = exp(e - max[dst]); segment sum ----------
__global__ void k_edge_expsum(const int* __restrict__ ei,
                              const float* __restrict__ as, const float* __restrict__ ad,
                              const unsigned* __restrict__ maxu,
                              float* __restrict__ denom, float* __restrict__ exbuf) {
  int e = blockIdx.x * blockDim.x + threadIdx.x;
  if (e >= ETOT) return;
  int s = (e < NEDGES) ? ei[e]          : (e - NEDGES);
  int d = (e < NEDGES) ? ei[NEDGES + e] : (e - NEDGES);
  const float4 vs = *(const float4*)(as + s * 4);
  const float4 vd = *(const float4*)(ad + d * 4);
  float v[4] = { vs.x + vd.x, vs.y + vd.y, vs.z + vd.z, vs.w + vd.w };
  float ex[4];
#pragma unroll
  for (int h = 0; h < 4; ++h) {
    float lv = v[h] > 0.f ? v[h] : SLOPE * v[h];
    ex[h] = __expf(lv - ord2f(maxu[d * 4 + h]));
    atomicAdd(&denom[d * 4 + h], ex[h]);
  }
  *(float4*)(exbuf + (size_t)e * 4) = make_float4(ex[0], ex[1], ex[2], ex[3]);
}

// ---------- edge pass 3: out[dst] += (ex/denom[dst]) * hlin[src]; 16 threads/(e,h) ----
__global__ void k_edge_agg(const int* __restrict__ ei,
                           const float* __restrict__ hlin,
                           const float* __restrict__ exbuf,
                           const float* __restrict__ denom,
                           float* __restrict__ out) {
  long long tid = (long long)blockIdx.x * blockDim.x + threadIdx.x;
  if (tid >= (long long)ETOT * 64) return;
  int e   = (int)(tid >> 6);
  int sub = (int)(tid & 63);
  int h   = sub >> 4;
  int c0  = (sub & 15) << 2;
  int s = (e < NEDGES) ? ei[e]          : (e - NEDGES);
  int d = (e < NEDGES) ? ei[NEDGES + e] : (e - NEDGES);
  float w = exbuf[(size_t)e * 4 + h] / denom[d * 4 + h];
  const float4 hv = *(const float4*)(hlin + (size_t)s * HC + h * CH + c0);
  float* o = out + (size_t)d * HC + h * CH + c0;
  atomicAdd(o + 0, w * hv.x);
  atomicAdd(o + 1, w * hv.y);
  atomicAdd(o + 2, w * hv.z);
  atomicAdd(o + 3, w * hv.w);
}

// ---------- bias + ELU in place ----------
__global__ void k_bias_elu(float* __restrict__ h, const float* __restrict__ b, int n) {
  long long i = (long long)blockIdx.x * blockDim.x + threadIdx.x;
  if (i >= (long long)n * HC) return;
  float v = h[i] + b[i & (HC - 1)];
  h[i] = v > 0.f ? v : (__expf(v) - 1.f);
}

// ---------- mean pool (sums + counts) ----------
__global__ void k_pool_add(const float* __restrict__ h, const int* __restrict__ batch,
                           float* __restrict__ pooled, int n) {
  long long i = (long long)blockIdx.x * blockDim.x + threadIdx.x;
  if (i >= (long long)n * HC) return;
  int node = (int)(i >> 8), j = (int)(i & 255);
  atomicAdd(&pooled[(size_t)batch[node] * HC + j], h[i]);
}
__global__ void k_pool_cnt(const int* __restrict__ batch, float* __restrict__ cnt, int n) {
  int i = blockIdx.x * blockDim.x + threadIdx.x;
  if (i < n) atomicAdd(&cnt[batch[i]], 1.0f);
}

// ---------- classifier head: logits[G,2] ----------
__global__ void k_head(const float* __restrict__ pooled, const float* __restrict__ cnt,
                       const float* __restrict__ wlin, const float* __restrict__ blin,
                       float* __restrict__ out) {
  int t = blockIdx.x * blockDim.x + threadIdx.x;
  if (t >= NGRAPH * 2) return;
  int g = t >> 1, j = t & 1;
  float inv = 1.0f / fmaxf(cnt[g], 1.0f);
  float acc = blin[j];
  const float* p = pooled + (size_t)g * HC;
#pragma unroll 4
  for (int c = 0; c < HC; ++c) acc += p[c] * inv * wlin[c * 2 + j];
  out[t] = acc;
}

static inline int nblk(long long n, int b) { return (int)((n + b - 1) / b); }

extern "C" void kernel_launch(void* const* d_in, const int* in_sizes, int n_in,
                              void* d_out, int out_size, void* d_ws, size_t ws_size,
                              hipStream_t stream) {
  const float* x       = (const float*)d_in[0];
  const int*   ei      = (const int*)d_in[1];
  const int*   batch   = (const int*)d_in[2];
  const float* W1      = (const float*)d_in[3];
  const float* a_src1  = (const float*)d_in[4];
  const float* a_dst1  = (const float*)d_in[5];
  const float* b1      = (const float*)d_in[6];
  const float* W2      = (const float*)d_in[7];
  const float* a_src2  = (const float*)d_in[8];
  const float* a_dst2  = (const float*)d_in[9];
  const float* b2      = (const float*)d_in[10];
  const float* W_lin   = (const float*)d_in[11];
  const float* b_lin   = (const float*)d_in[12];
  float* out = (float*)d_out;
  (void)in_sizes; (void)n_in; (void)out_size; (void)ws_size;

  // ---- carve workspace (256B aligned regions) ----
  char* base = (char*)d_ws;
  size_t off = 0;
  auto carve = [&](size_t bytes) -> void* {
    void* p = base + off;
    off += (bytes + 255) & ~(size_t)255;
    return p;
  };
  _Float16* Ah    = (_Float16*)carve((size_t)NNODES * HC * sizeof(_Float16)); // 25.6 MB
  _Float16* Bp    = (_Float16*)carve((size_t)16 * 8 * 512 * sizeof(_Float16));// 128 KB
  float*    hlin  = (float*)carve((size_t)NNODES * HC * 4);                   // 51.2 MB
  float*    hout  = (float*)carve((size_t)NNODES * HC * 4);                   // 51.2 MB
  float*    as    = (float*)carve((size_t)NNODES * HEADS * 4);
  float*    ad    = (float*)carve((size_t)NNODES * HEADS * 4);
  unsigned* maxu  = (unsigned*)carve((size_t)NNODES * HEADS * 4);
  float*    denom = (float*)carve((size_t)NNODES * HEADS * 4);
  float*    exbuf = (float*)carve((size_t)ETOT * HEADS * 4);                  // 13.6 MB
  float*    pooled= (float*)carve((size_t)NGRAPH * HC * 4);
  float*    cnt   = (float*)carve((size_t)NGRAPH * 4);

  const int B = 256;
  const long long nHC   = (long long)NNODES * HC;
  const long long aggN  = (long long)ETOT * 64;
  const int gemmBlocks  = nblk((long long)(NNODES / 16) * (HC / 16) * 32, B);

  // ================= layer 1 =================
  k_f32_to_f16<<<nblk((long long)NNODES * INC, B), B, 0, stream>>>(Ah, x, (long long)NNODES * INC);
  k_pack_w<<<nblk((HC / 16) * (INC / 32) * 512, B), B, 0, stream>>>(Bp, W1, INC, HC);
  k_gemm_wmma<<<gemmBlocks, B, 0, stream>>>(Ah, Bp, hlin, NNODES, INC, HC);
  k_alphas<<<nblk((long long)NNODES * HEADS, B), B, 0, stream>>>(hlin, a_src1, a_dst1, as, ad, NNODES);
  k_fill_u32<<<nblk((long long)NNODES * HEADS, B), B, 0, stream>>>(maxu, 0u, (long long)NNODES * HEADS);
  k_fill_u32<<<nblk((long long)NNODES * HEADS, B), B, 0, stream>>>((unsigned*)denom, 0u, (long long)NNODES * HEADS);
  k_fill_u32<<<nblk(nHC, B), B, 0, stream>>>((unsigned*)hout, 0u, nHC);
  k_edge_max<<<nblk(ETOT, B), B, 0, stream>>>(ei, as, ad, maxu);
  k_edge_expsum<<<nblk(ETOT, B), B, 0, stream>>>(ei, as, ad, maxu, denom, exbuf);
  k_edge_agg<<<nblk(aggN, B), B, 0, stream>>>(ei, hlin, exbuf, denom, hout);
  k_bias_elu<<<nblk(nHC, B), B, 0, stream>>>(hout, b1, NNODES);

  // ================= layer 2 =================
  k_f32_to_f16<<<nblk(nHC, B), B, 0, stream>>>(Ah, hout, nHC);
  k_pack_w<<<nblk((HC / 16) * (HC / 32) * 512, B), B, 0, stream>>>(Bp, W2, HC, HC);
  k_gemm_wmma<<<gemmBlocks, B, 0, stream>>>(Ah, Bp, hlin, NNODES, HC, HC);
  k_alphas<<<nblk((long long)NNODES * HEADS, B), B, 0, stream>>>(hlin, a_src2, a_dst2, as, ad, NNODES);
  k_fill_u32<<<nblk((long long)NNODES * HEADS, B), B, 0, stream>>>(maxu, 0u, (long long)NNODES * HEADS);
  k_fill_u32<<<nblk((long long)NNODES * HEADS, B), B, 0, stream>>>((unsigned*)denom, 0u, (long long)NNODES * HEADS);
  k_fill_u32<<<nblk(nHC, B), B, 0, stream>>>((unsigned*)hout, 0u, nHC);
  k_edge_max<<<nblk(ETOT, B), B, 0, stream>>>(ei, as, ad, maxu);
  k_edge_expsum<<<nblk(ETOT, B), B, 0, stream>>>(ei, as, ad, maxu, denom, exbuf);
  k_edge_agg<<<nblk(aggN, B), B, 0, stream>>>(ei, hlin, exbuf, denom, hout);
  k_bias_elu<<<nblk(nHC, B), B, 0, stream>>>(hout, b2, NNODES);

  // ================= pool + head =================
  k_fill_u32<<<nblk((long long)NGRAPH * HC, B), B, 0, stream>>>((unsigned*)pooled, 0u, (long long)NGRAPH * HC);
  k_fill_u32<<<1, B, 0, stream>>>((unsigned*)cnt, 0u, NGRAPH);
  k_pool_add<<<nblk(nHC, B), B, 0, stream>>>(hout, batch, pooled, NNODES);
  k_pool_cnt<<<nblk(NNODES, B), B, 0, stream>>>(batch, cnt, NNODES);
  k_head<<<1, 128, 0, stream>>>(pooled, cnt, W_lin, b_lin, out);
}
